// TransformerLayer_43628277792905
// MI455X (gfx1250) — compile-verified, matrix operand order checked by reading, not run on hardware
//
#include <hip/hip_runtime.h>
#include <cstddef>
#include <cstdint>

typedef unsigned short u16;
typedef __attribute__((ext_vector_type(16))) __bf16 v16bf;
typedef __attribute__((ext_vector_type(8)))  __bf16 v8bf;
typedef __attribute__((ext_vector_type(8)))  float  v8f;

union FragBF { v16bf v; v8bf h[2]; };

__device__ __forceinline__ u16 f2bf(float x) {
  unsigned u = __float_as_uint(x);
  u += 0x7FFFu + ((u >> 16) & 1u);   // round-to-nearest-even
  return (u16)(u >> 16);
}

__device__ __forceinline__ v8f wmma_bf16(const FragBF& a, const FragBF& b, v8f c) {
  return __builtin_amdgcn_wmma_f32_16x16x32_bf16(false, a.v, false, b.v,
                                                 (short)0, c, false, false);
}

#define V8F_ZERO v8f{0.f,0.f,0.f,0.f,0.f,0.f,0.f,0.f}

// ---------------------------------------------------------------------------
// fp32 -> bf16 elementwise convert (weights)
// ---------------------------------------------------------------------------
__global__ __launch_bounds__(256) void f32_to_bf16_kernel(
    const float* __restrict__ in, u16* __restrict__ out, int n) {
  int i = blockIdx.x * 256 + threadIdx.x;
  if (i < n) out[i] = f2bf(in[i]);
}

// ---------------------------------------------------------------------------
// LayerNorm (row = 1024 fp32) -> bf16
// ---------------------------------------------------------------------------
__global__ __launch_bounds__(256) void layernorm_bf16_kernel(
    const float* __restrict__ x, const float* __restrict__ g,
    const float* __restrict__ b, u16* __restrict__ out, int D) {
  const int row = blockIdx.x;
  const float* xr = x + (size_t)row * D;
  float s = 0.f, s2 = 0.f;
  for (int i = threadIdx.x; i < D; i += 256) { float v = xr[i]; s += v; s2 += v * v; }
  __shared__ float red[256], red2[256];
  red[threadIdx.x] = s; red2[threadIdx.x] = s2;
  __syncthreads();
  for (int off = 128; off > 0; off >>= 1) {
    if (threadIdx.x < off) {
      red[threadIdx.x]  += red[threadIdx.x + off];
      red2[threadIdx.x] += red2[threadIdx.x + off];
    }
    __syncthreads();
  }
  const float mu  = red[0] / (float)D;
  const float var = red2[0] / (float)D - mu * mu;
  const float rstd = rsqrtf(var + 1e-6f);
  u16* orow = out + (size_t)row * D;
  for (int i = threadIdx.x; i < D; i += 256)
    orow[i] = f2bf((xr[i] - mu) * rstd * g[i] + b[i]);
}

// ---------------------------------------------------------------------------
// Tiled bf16 WMMA GEMM: C[M,N] = A[M,K] * B[K,N] (+bias, +gelu, +residual)
// 128x128 block tile, 256 threads (8 waves in 2x4 grid, each wave 64x32).
// Double-buffered LDS: one barrier per K-step, next-tile global loads overlap
// the current tile's WMMA chain; prefetch t+2 tiles into cache.
// ---------------------------------------------------------------------------
template <bool BIAS, bool GELU, bool RES, bool OUTF32>
__global__ __launch_bounds__(256) void gemm_bf16_kernel(
    const u16* __restrict__ A, const u16* __restrict__ Bw,
    const float* __restrict__ bias, const float* __restrict__ res,
    void* __restrict__ outp, int M, int N, int K) {
  __shared__ __align__(16) u16 As[2][128][40];   // [buf][m][k], +8 pad
  __shared__ __align__(16) u16 Bs[2][128][40];   // B^T: [buf][n][k], +8 pad
  const int tid  = threadIdx.x;
  const int lane = tid & 31, wave = tid >> 5;
  const int wr = wave >> 2, wc = wave & 3;
  const int hsel = lane >> 4, l16 = lane & 15;
  const int n0 = blockIdx.x * 128, m0 = blockIdx.y * 128;

  // per-thread chunk coordinates (2 chunks each for A and B per tile)
  int rA[2], kA[2], kB[2], nB[2];
#pragma unroll
  for (int i = 0; i < 2; i++) {
    const int c = tid + 256 * i;
    rA[i] = c >> 2;        kA[i] = (c & 3) * 8;    // A: 128 rows x 4 chunks
    kB[i] = c >> 4;        nB[i] = (c & 15) * 8;   // B: 32 rows x 16 chunks
  }

  auto load_tile = [&](int k0, uint4 (&ar)[2], uint4 (&br)[2]) {
#pragma unroll
    for (int i = 0; i < 2; i++) {
      ar[i] = *(const uint4*)(A  + (size_t)(m0 + rA[i]) * K + k0 + kA[i]);
      br[i] = *(const uint4*)(Bw + (size_t)(k0 + kB[i]) * N + n0 + nB[i]);
    }
  };
  auto store_tile = [&](int buf, const uint4 (&ar)[2], const uint4 (&br)[2]) {
#pragma unroll
    for (int i = 0; i < 2; i++) {
      *(uint4*)&As[buf][rA[i]][kA[i]] = ar[i];
      u16 tmp[8]; *(uint4*)tmp = br[i];
#pragma unroll
      for (int j = 0; j < 8; j++) Bs[buf][nB[i] + j][kB[i]] = tmp[j];  // transpose
    }
  };

  v8f acc[4][2];
#pragma unroll
  for (int i = 0; i < 4; i++)
#pragma unroll
    for (int j = 0; j < 2; j++) acc[i][j] = V8F_ZERO;

  const int T = K >> 5;  // number of 32-wide K-steps
  {
    uint4 ar[2], br[2];
    load_tile(0, ar, br);
    store_tile(0, ar, br);
  }
  __syncthreads();

  for (int t = 0; t < T; t++) {
    const int cur = t & 1;
    if (t + 1 < T) {
      // overlap: fetch tile t+1 and stage into the other buffer while the
      // WMMA chain below consumes buffer `cur` (safe: buf 1-cur readers
      // finished before the barrier that ended iteration t-1)
      uint4 ar[2], br[2];
      load_tile((t + 1) * 32, ar, br);
      store_tile(1 - cur, ar, br);
      if (t + 2 < T) {
        __builtin_prefetch(A  + (size_t)(m0 + rA[0]) * K + (t + 2) * 32 + kA[0], 0, 1);
        __builtin_prefetch(Bw + (size_t)((t + 2) * 32 + kB[0]) * N + n0 + nB[0], 0, 1);
      }
    }

    FragBF bfr[2];
#pragma unroll
    for (int ni = 0; ni < 2; ni++) {
      const u16* p = &Bs[cur][wc * 32 + ni * 16 + l16][0];
      bfr[ni].h[0] = *(const v8bf*)(p + 8 * hsel);
      bfr[ni].h[1] = *(const v8bf*)(p + 16 + 8 * hsel);
    }
#pragma unroll
    for (int mi = 0; mi < 4; mi++) {
      const u16* p = &As[cur][wr * 64 + mi * 16 + l16][0];
      FragBF af;
      af.h[0] = *(const v8bf*)(p + 8 * hsel);
      af.h[1] = *(const v8bf*)(p + 16 + 8 * hsel);
#pragma unroll
      for (int ni = 0; ni < 2; ni++)
        acc[mi][ni] = wmma_bf16(af, bfr[ni], acc[mi][ni]);
    }
    __syncthreads();
  }

  // epilogue: C layout -> m = 8*hsel + r, n = l16 within each 16x16 fragment
#pragma unroll
  for (int mi = 0; mi < 4; mi++) {
#pragma unroll
    for (int ni = 0; ni < 2; ni++) {
      const int n = n0 + wc * 32 + ni * 16 + l16;
      const float bv = BIAS ? bias[n] : 0.0f;
#pragma unroll
      for (int r = 0; r < 8; r++) {
        const int m = m0 + wr * 64 + mi * 16 + hsel * 8 + r;
        float v = acc[mi][ni][r] + bv;
        if (GELU) {
          float t = v + 0.044715f * v * v * v;
          v = 0.5f * v * (1.0f + tanhf(0.7978845608028654f * t));
        }
        const size_t idx = (size_t)m * N + n;
        if (RES) v += res[idx];
        if (OUTF32) ((float*)outp)[idx] = v;
        else        ((u16*)outp)[idx]   = f2bf(v);
      }
    }
  }
}

// ---------------------------------------------------------------------------
// Flash attention: one workgroup per (batch, head, 128-query block).
// Q,K,V are bf16 [B*S, 1024] with head h at columns h*64..h*64+63.
// ---------------------------------------------------------------------------
constexpr int ATT_SMEM_BYTES =
    (128 * 72 + 128 * 72 + 64 * 136) * 2 + (128 * 128 + 3 * 128) * 4; // 121344

__global__ __launch_bounds__(256) void attention_kernel(
    const u16* __restrict__ Q, const u16* __restrict__ Kmat,
    const u16* __restrict__ V, u16* __restrict__ O, int S, int D) {
  extern __shared__ __align__(16) char smem[];
  u16*   Qs   = (u16*)smem;                 // 128 x 72 (64 + pad)
  u16*   Ks   = Qs + 128 * 72;              // 128 x 72
  u16*   Vts  = Ks + 128 * 72;              // V^T: 64 x 136 (128 + pad)
  float* Ss   = (float*)(Vts + 64 * 136);   // 128 x 128 f32; P(bf16) overlaid per-row
  float* mrun = Ss + 128 * 128;
  float* lrun = mrun + 128;
  float* fac  = lrun + 128;

  const int qb = blockIdx.x, hd = blockIdx.y, bb = blockIdx.z;
  const int tid = threadIdx.x, lane = tid & 31, wave = tid >> 5;
  const int hsel = lane >> 4, l16 = lane & 15;
  const int qm0 = qb * 128;
  const size_t base = (size_t)bb * S * D;
  const int coff = hd * 64;

  // load Q tile 128x64
#pragma unroll
  for (int i = 0; i < 4; i++) {
    int c = tid + 256 * i;
    int r = c >> 3, kc = (c & 7) * 8;
    uint4 v = *(const uint4*)(Q + base + (size_t)(qm0 + r) * D + coff + kc);
    *(uint4*)&Qs[r * 72 + kc] = v;
  }
  if (tid < 128) { mrun[tid] = -1e30f; lrun[tid] = 0.0f; }

  const int owr = wave >> 1, owc = wave & 1;  // O: 4x2 wave grid (32x32 each)
  const int swr = wave >> 2, swc = wave & 3;  // S: 2x4 wave grid (64x32 each)

  v8f oacc[2][2];
#pragma unroll
  for (int i = 0; i < 2; i++)
#pragma unroll
    for (int j = 0; j < 2; j++) oacc[i][j] = V8F_ZERO;

  for (int kt = 0; kt <= qb; kt++) {
    const int kt0 = kt * 128;
    __syncthreads();
    // K tile 128x64 and V tile (transposed) 64x128
#pragma unroll
    for (int i = 0; i < 4; i++) {
      int c = tid + 256 * i;
      int r = c >> 3, kc = (c & 7) * 8;
      uint4 v = *(const uint4*)(Kmat + base + (size_t)(kt0 + r) * D + coff + kc);
      *(uint4*)&Ks[r * 72 + kc] = v;
    }
#pragma unroll
    for (int i = 0; i < 4; i++) {
      int c = tid + 256 * i;
      int r = c >> 3, dc = (c & 7) * 8;
      uint4 v = *(const uint4*)(V + base + (size_t)(kt0 + r) * D + coff + dc);
      u16 tmp[8]; *(uint4*)tmp = v;
#pragma unroll
      for (int j = 0; j < 8; j++) Vts[(dc + j) * 136 + r] = tmp[j];
    }
    // prefetch next key/value tiles into cache while this tile is consumed
    if (kt < qb) {
#pragma unroll
      for (int i = 0; i < 4; i++) {
        int c = tid + 256 * i;
        int r = c >> 3, kc = (c & 7) * 8;
        __builtin_prefetch(Kmat + base + (size_t)(kt0 + 128 + r) * D + coff + kc, 0, 1);
        __builtin_prefetch(V    + base + (size_t)(kt0 + 128 + r) * D + coff + kc, 0, 1);
      }
    }
    __syncthreads();

    // scores = Q * K^T  (K-fragment gather == B^T layout == Ks row-major)
    v8f sacc[4][2];
#pragma unroll
    for (int i = 0; i < 4; i++)
#pragma unroll
      for (int j = 0; j < 2; j++) sacc[i][j] = V8F_ZERO;
    for (int ks = 0; ks < 64; ks += 32) {
      FragBF bfr[2];
#pragma unroll
      for (int ni = 0; ni < 2; ni++) {
        const u16* p = &Ks[(swc * 32 + ni * 16 + l16) * 72 + ks];
        bfr[ni].h[0] = *(const v8bf*)(p + 8 * hsel);
        bfr[ni].h[1] = *(const v8bf*)(p + 16 + 8 * hsel);
      }
#pragma unroll
      for (int mi = 0; mi < 4; mi++) {
        const u16* p = &Qs[(swr * 64 + mi * 16 + l16) * 72 + ks];
        FragBF af;
        af.h[0] = *(const v8bf*)(p + 8 * hsel);
        af.h[1] = *(const v8bf*)(p + 16 + 8 * hsel);
#pragma unroll
        for (int ni = 0; ni < 2; ni++)
          sacc[mi][ni] = wmma_bf16(af, bfr[ni], sacc[mi][ni]);
      }
    }
    // spill scores to LDS
#pragma unroll
    for (int mi = 0; mi < 4; mi++)
#pragma unroll
      for (int ni = 0; ni < 2; ni++) {
        const int n = swc * 32 + ni * 16 + l16;
#pragma unroll
        for (int r = 0; r < 8; r++) {
          const int m = swr * 64 + mi * 16 + hsel * 8 + r;
          Ss[m * 128 + n] = sacc[mi][ni][r];
        }
      }
    __syncthreads();

    // per-row online softmax (causal mask on diagonal tile)
    if (tid < 128) {
      float* srow = Ss + tid * 128;
      u16*   prow = (u16*)srow;                 // bf16 P overlays the row
      const int lim = (kt == qb) ? (tid + 1) : 128;
      float mx = -1e30f;
      for (int j = 0; j < lim; j++) {
        float s = srow[j] * 0.125f;             // 1/sqrt(64)
        srow[j] = s;
        mx = s > mx ? s : mx;
      }
      const float mold = mrun[tid];
      const float mnew = mx > mold ? mx : mold;
      const float f = __expf(mold - mnew);
      float lsum = 0.0f;
      for (int j = 0; j < lim; j++) {
        float p = __expf(srow[j] - mnew);
        lsum += p;
        prow[j] = f2bf(p);                      // writes slot j/2 <= already-read
      }
      for (int j = lim; j < 128; j++) prow[j] = 0;
      fac[tid]  = f;
      mrun[tid] = mnew;
      lrun[tid] = lrun[tid] * f + lsum;
    }
    __syncthreads();

    // O = O*fac + P * V   (V-fragment gather uses Vts = V^T)
#pragma unroll
    for (int mi = 0; mi < 2; mi++)
#pragma unroll
      for (int r = 0; r < 8; r++) {
        const float f = fac[owr * 32 + mi * 16 + hsel * 8 + r];
        oacc[mi][0][r] *= f;
        oacc[mi][1][r] *= f;
      }
    for (int ks = 0; ks < 128; ks += 32) {
      FragBF bfr[2];
#pragma unroll
      for (int ni = 0; ni < 2; ni++) {
        const u16* p = &Vts[(owc * 32 + ni * 16 + l16) * 136 + ks];
        bfr[ni].h[0] = *(const v8bf*)(p + 8 * hsel);
        bfr[ni].h[1] = *(const v8bf*)(p + 16 + 8 * hsel);
      }
#pragma unroll
      for (int mi = 0; mi < 2; mi++) {
        const u16* p = (const u16*)(Ss + (owr * 32 + mi * 16 + l16) * 128) + ks;
        FragBF af;
        af.h[0] = *(const v8bf*)(p + 8 * hsel);
        af.h[1] = *(const v8bf*)(p + 16 + 8 * hsel);
#pragma unroll
        for (int ni = 0; ni < 2; ni++)
          oacc[mi][ni] = wmma_bf16(af, bfr[ni], oacc[mi][ni]);
      }
    }
  }

  // normalize and store
#pragma unroll
  for (int mi = 0; mi < 2; mi++)
#pragma unroll
    for (int ni = 0; ni < 2; ni++) {
      const int n = owc * 32 + ni * 16 + l16;
#pragma unroll
      for (int r = 0; r < 8; r++) {
        const int m = owr * 32 + mi * 16 + hsel * 8 + r;
        const float inv = 1.0f / lrun[m];
        O[base + (size_t)(qm0 + m) * D + coff + n] = f2bf(oacc[mi][ni][r] * inv);
      }
    }
}

// ---------------------------------------------------------------------------
// host orchestration
// ---------------------------------------------------------------------------
extern "C" void kernel_launch(void* const* d_in, const int* in_sizes, int n_in,
                              void* d_out, int out_size, void* d_ws, size_t ws_size,
                              hipStream_t stream) {
  (void)in_sizes; (void)n_in; (void)out_size; (void)ws_size;
  const int B = 4, S = 2048, D = 1024, F = 4096;
  const int M = B * S;  // 8192

  const float* x     = (const float*)d_in[0];
  const float* Wq    = (const float*)d_in[1];
  const float* bq    = (const float*)d_in[2];
  const float* Wk    = (const float*)d_in[3];
  const float* bk    = (const float*)d_in[4];
  const float* Wv    = (const float*)d_in[5];
  const float* bv    = (const float*)d_in[6];
  const float* Wo    = (const float*)d_in[7];
  const float* W1    = (const float*)d_in[8];
  const float* b1    = (const float*)d_in[9];
  const float* W2    = (const float*)d_in[10];
  const float* b2    = (const float*)d_in[11];
  const float* g1    = (const float*)d_in[12];
  const float* beta1 = (const float*)d_in[13];
  const float* g2    = (const float*)d_in[14];
  const float* beta2 = (const float*)d_in[15];

  char* ws = (char*)d_ws;
  size_t off = 0;
  auto alloc = [&](size_t bytes) -> void* {
    void* p = ws + off;
    off += (bytes + 255) & ~(size_t)255;
    return p;
  };
  u16* wq_b = (u16*)alloc((size_t)D * D * 2);
  u16* wk_b = (u16*)alloc((size_t)D * D * 2);
  u16* wv_b = (u16*)alloc((size_t)D * D * 2);
  u16* wo_b = (u16*)alloc((size_t)D * D * 2);
  u16* w1_b = (u16*)alloc((size_t)D * F * 2);
  u16* w2_b = (u16*)alloc((size_t)F * D * 2);
  u16* xn1  = (u16*)alloc((size_t)M * D * 2);
  u16* qb_  = (u16*)alloc((size_t)M * D * 2);
  u16* kb_  = (u16*)alloc((size_t)M * D * 2);
  u16* vb_  = (u16*)alloc((size_t)M * D * 2);
  u16* attn = (u16*)alloc((size_t)M * D * 2);
  float* x1 = (float*)alloc((size_t)M * D * 4);
  u16* xn2  = (u16*)alloc((size_t)M * D * 2);
  u16* hbuf = (u16*)alloc((size_t)M * F * 2);

  // weight conversion
  f32_to_bf16_kernel<<<(D * D + 255) / 256, 256, 0, stream>>>(Wq, wq_b, D * D);
  f32_to_bf16_kernel<<<(D * D + 255) / 256, 256, 0, stream>>>(Wk, wk_b, D * D);
  f32_to_bf16_kernel<<<(D * D + 255) / 256, 256, 0, stream>>>(Wv, wv_b, D * D);
  f32_to_bf16_kernel<<<(D * D + 255) / 256, 256, 0, stream>>>(Wo, wo_b, D * D);
  f32_to_bf16_kernel<<<(D * F + 255) / 256, 256, 0, stream>>>(W1, w1_b, D * F);
  f32_to_bf16_kernel<<<(F * D + 255) / 256, 256, 0, stream>>>(W2, w2_b, F * D);

  // LN1
  layernorm_bf16_kernel<<<M, 256, 0, stream>>>(x, g1, beta1, xn1, D);

  // QKV projections (bias, bf16 out)
  dim3 gDD(D / 128, M / 128);
  gemm_bf16_kernel<true, false, false, false><<<gDD, 256, 0, stream>>>(
      xn1, wq_b, bq, nullptr, qb_, M, D, D);
  gemm_bf16_kernel<true, false, false, false><<<gDD, 256, 0, stream>>>(
      xn1, wk_b, bk, nullptr, kb_, M, D, D);
  gemm_bf16_kernel<true, false, false, false><<<gDD, 256, 0, stream>>>(
      xn1, wv_b, bv, nullptr, vb_, M, D, D);

  // flash attention
  hipFuncSetAttribute((const void*)attention_kernel,
                      hipFuncAttributeMaxDynamicSharedMemorySize, ATT_SMEM_BYTES);
  attention_kernel<<<dim3(S / 128, 16, B), 256, ATT_SMEM_BYTES, stream>>>(
      qb_, kb_, vb_, attn, S, D);

  // O projection + residual(x) -> x1 (fp32)
  gemm_bf16_kernel<false, false, true, true><<<gDD, 256, 0, stream>>>(
      attn, wo_b, nullptr, x, x1, M, D, D);

  // LN2
  layernorm_bf16_kernel<<<M, 256, 0, stream>>>(x1, g2, beta2, xn2, D);

  // FFN1: gelu(xn2 @ W1 + b1) -> bf16
  gemm_bf16_kernel<true, true, false, false><<<dim3(F / 128, M / 128), 256, 0, stream>>>(
      xn2, w1_b, b1, nullptr, hbuf, M, F, D);

  // FFN2: h @ W2 + b2 + x1 -> d_out (fp32)
  gemm_bf16_kernel<true, false, true, true><<<gDD, 256, 0, stream>>>(
      hbuf, w2_b, b2, x1, (float*)d_out, M, D, F);
}
